// S2VTModel_41970420417756
// MI455X (gfx1250) — matrix operand order repaired and checked by reading, not compiled, not run on hardware
//
#include <hip/hip_runtime.h>
#include <hip/hip_bf16.h>
#include <math.h>

// ---------------------------------------------------------------------------
// S2VT (2-layer GRU video captioner) for MI455X / gfx1250, wave32 + WMMA.
//
// GEMMs: Y[M,N] = X[M,K] @ W[N,K]^T (+bias / +=Y) on v_wmma_f32_16x16x32_f16.
// Fast path (if ws_size allows ~118MB): weights/activations pre-converted to
// f16; B (weight) tiles staged into LDS by the Tensor Data Mover
// (tensor_load_to_lds, double-buffered, TENSORcnt-tracked) with D# padding
// producing a bank-conflict-free 40-half row stride. Fallback path converts
// f32->f16 on the fly while staging through VGPRs.
// The 27 decoder vocab projections are batched into ONE 1728x32000x1024 GEMM.
// ---------------------------------------------------------------------------

typedef __attribute__((ext_vector_type(16))) _Float16 v16h;
typedef __attribute__((ext_vector_type(8)))  float    v8f;
typedef unsigned int u32;
typedef __attribute__((ext_vector_type(4))) u32 u32x4;
typedef __attribute__((ext_vector_type(4))) int i32x4;
typedef __attribute__((ext_vector_type(8))) int i32x8;

#define GRU_H   1024
#define GRU_3H  3072
#define TDEC    27
#define VOCAB   32000
#define LDF     40          // LDS row stride in halves (80B: 16B-aligned, conflict-free)

#define FLAG_ACC    1       // Y += result
#define FLAG_REMAP  2       // row m = t*64+b -> row b*27+t (logits into d_out)

#if __has_builtin(__builtin_amdgcn_tensor_load_to_lds)
#define HAVE_TDM 1
#else
#define HAVE_TDM 0
#endif

union FragH { uint4 u[2]; v16h h; };

#if HAVE_TDM
// Issue one TDM 2-D tile load: tile_rows x 32 f16 elements from a row-major
// (ldw-elem stride) f16 matrix into LDS at lds_off, padding +4 DWORDs every
// 16 DWORDs => LDS rows of LDF=40 halves.  (ISA 08_async_tensor D# layout.)
__device__ __forceinline__ void tdm_load_b_tile(const _Float16* gsrc, u32 lds_off,
                                                int tile_rows, long tdim0, long tdim1,
                                                long ldw)
{
    unsigned long long ga = (unsigned long long)(size_t)gsrc;
    u32x4 g0;
    g0[0] = 1u;                                             // count=1, user mode
    g0[1] = lds_off;                                        // lds_addr (bytes)
    g0[2] = (u32)(ga & 0xFFFFFFFFull);                      // global_addr[31:0]
    g0[3] = (u32)((ga >> 32) & 0x01FFFFFFull) | (2u << 30); // addr[56:32] | type=2
    i32x8 g1;
    g1[0] = (1 << 16) | (1 << 20) | (3 << 22) | (3 << 25);  // 2B elems, pad 16DW->+4DW
    g1[1] = (int)((tdim0 & 0xFFFF) << 16);                  // tensor_dim0[15:0]
    g1[2] = (int)(((tdim0 >> 16) & 0xFFFF) | ((tdim1 & 0xFFFF) << 16));
    g1[3] = (int)(((tdim1 >> 16) & 0xFFFF) | (32u << 16));  // tile_dim0 = 32
    g1[4] = tile_rows & 0xFFFF;                             // tile_dim1 (dim2=0)
    g1[5] = (int)(ldw & 0xFFFFFFFFll);                      // dim0_stride lo32
    g1[6] = (int)((ldw >> 32) & 0xFFFF);                    // dim0_stride hi16
    g1[7] = 0;
    i32x4 z4 = {0, 0, 0, 0};
#if __has_include(<hip/amd_detail/amd_gfx1250_TDM.h>)
    i32x8 z8 = {0, 0, 0, 0, 0, 0, 0, 0};
    __builtin_amdgcn_tensor_load_to_lds(g0, g1, z4, z4, z8, 0);   // clang-23 arity
#else
    __builtin_amdgcn_tensor_load_to_lds(g0, g1, z4, z4, 0);       // ROCm 7.2 arity
#endif
}
#endif

// ---------------------------------------------------------------------------
// f16-direct WMMA GEMM.  grid = (N/(32*NSUB), M/64), block = 256 (8 waves).
// Wave owns NSUB 16x16 C tiles.  B staged by TDM (double buffer), A by b128
// LDS copies.  K in chunks of 32.
// ---------------------------------------------------------------------------
template <int NSUB>
__global__ __launch_bounds__(256) void gemm16(
    const _Float16* __restrict__ X, int ldx,
    const _Float16* __restrict__ W, int ldw,
    const float* __restrict__ bias,
    float* __restrict__ Y, int N, int K, int flags)
{
    constexpr int BN = 32 * NSUB;
    __shared__ __align__(16) _Float16 As[64 * LDF];
    __shared__ __align__(16) _Float16 Bs[2][BN * LDF];

    const int tid  = threadIdx.x;
    const int lane = tid & 31;
    const int wave = tid >> 5;
    const int m0   = blockIdx.y * 64;
    const int n0   = blockIdx.x * BN;
    const int mi   = wave & 3;
    const int ni   = wave >> 2;

    const int arow = tid >> 2;
    const int akg  = (tid & 3) * 8;
    const _Float16* gA = X + (size_t)(m0 + arow) * (size_t)ldx + akg;
    const _Float16* gW = W + (size_t)n0 * (size_t)ldw;

    const int nk = K >> 5;
#if HAVE_TDM
    if (wave == 0)
        tdm_load_b_tile(gW, (u32)(size_t)&Bs[0][0], BN, K, N, ldw);
#endif

    v8f c[NSUB];
#pragma unroll
    for (int s = 0; s < NSUB; ++s) c[s] = (v8f){};

    for (int kc = 0; kc < nk; ++kc) {
        const int s = kc & 1;
        // ---- stage A chunk (one 128-bit copy per thread) ----
        *(uint4*)&As[arow * LDF + akg] = *(const uint4*)(gA + ((size_t)kc << 5));
        if (kc + 1 < nk)
            __builtin_prefetch(gA + ((size_t)(kc + 1) << 5), 0, 3);
#if HAVE_TDM
        if (wave == 0) {
            if (kc + 1 < nk) {
                tdm_load_b_tile(gW + ((size_t)(kc + 1) << 5),
                                (u32)(size_t)&Bs[s ^ 1][0], BN, K, N, ldw);
                __builtin_amdgcn_s_wait_tensorcnt(1);   // chunk kc resident
            } else {
                __builtin_amdgcn_s_wait_tensorcnt(0);
            }
        }
#else
        {   // plain staging fallback
            constexpr int per = 256 / BN;           // threads per row
            constexpr int kpt = 32 / per;           // halves per thread
            const int brow = tid / per;
            const int bkg  = (tid % per) * kpt;
            const _Float16* src = gW + (size_t)brow * ldw + ((size_t)kc << 5) + bkg;
#pragma unroll
            for (int i = 0; i < kpt; ++i)
                Bs[s][brow * LDF + bkg + i] = src[i];
        }
#endif
        __syncthreads();

        // ---- A fragment: lane<16: M=lane, K=0..15; lane>=16: K=16..31
        FragH fa;
        {
            const int row = mi * 16 + (lane & 15);
            const int kb  = (lane < 16) ? 0 : 16;
            fa.u[0] = *(const uint4*)&As[row * LDF + kb];
            fa.u[1] = *(const uint4*)&As[row * LDF + kb + 8];
        }
#pragma unroll
        for (int sub = 0; sub < NSUB; ++sub) {
            // ---- B fragment: lane<16: N=lane, K=0..15; lane>=16: K=16..31
            FragH fb;
            const int row = (ni * NSUB + sub) * 16 + (lane & 15);
            const int kb  = (lane < 16) ? 0 : 16;
            fb.u[0] = *(const uint4*)&Bs[s][row * LDF + kb];
            fb.u[1] = *(const uint4*)&Bs[s][row * LDF + kb + 8];
            c[sub] = __builtin_amdgcn_wmma_f32_16x16x32_f16(false, fa.h, false, fb.h,
                                                            (short)0, c[sub], false, false);
        }
        __syncthreads();
    }

    // ---- writeback: C VGPR r holds M=r (lanes 0-15) / M=8+r (lanes 16-31), N=lane&15
    const int gm_base = m0 + mi * 16 + ((lane < 16) ? 0 : 8);
#pragma unroll
    for (int sub = 0; sub < NSUB; ++sub) {
        const int gn = n0 + (ni * NSUB + sub) * 16 + (lane & 15);
        const float badd = bias ? bias[gn] : 0.0f;
#pragma unroll
        for (int r = 0; r < 8; ++r) {
            const int gm = gm_base + r;
            size_t orow = (flags & FLAG_REMAP)
                            ? (size_t)((gm & 63) * TDEC + (gm >> 6))
                            : (size_t)gm;
            size_t off = orow * (size_t)N + (size_t)gn;
            float v = c[sub][r] + badd;
            if (flags & FLAG_ACC) v += Y[off];
            Y[off] = v;
        }
    }
}

// ---------------------------------------------------------------------------
// Fallback GEMM: f32 operands converted to f16 while staging (round-1 path).
// ---------------------------------------------------------------------------
__global__ __launch_bounds__(256) void gemm_xwT(
    const float* __restrict__ X, int ldx,
    const float* __restrict__ W, int ldw,
    const float* __restrict__ bias,
    float* __restrict__ Y, int N, int K, int flags)
{
    __shared__ __align__(16) _Float16 As[64 * LDF];
    __shared__ __align__(16) _Float16 Bs[32 * LDF];

    const int tid  = threadIdx.x;
    const int lane = tid & 31;
    const int wave = tid >> 5;
    const int m0   = blockIdx.y * 64;
    const int n0   = blockIdx.x * 32;
    const int mi   = wave & 3;
    const int ni   = wave >> 2;

    const int arow = tid >> 2;
    const int akg  = (tid & 3) * 8;
    const int brow = tid >> 3;
    const int bkg  = (tid & 7) * 4;

    const float* gA = X + (size_t)(m0 + arow) * (size_t)ldx + akg;
    const float* gB = W + (size_t)(n0 + brow) * (size_t)ldw + bkg;

    v8f c = {};
    const int nk = K >> 5;
    for (int kc = 0; kc < nk; ++kc) {
        const int k0 = kc << 5;
        float4 a0 = *(const float4*)(gA + k0);
        float4 a1 = *(const float4*)(gA + k0 + 4);
        _Float16* pa = &As[arow * LDF + akg];
        pa[0] = (_Float16)a0.x; pa[1] = (_Float16)a0.y;
        pa[2] = (_Float16)a0.z; pa[3] = (_Float16)a0.w;
        pa[4] = (_Float16)a1.x; pa[5] = (_Float16)a1.y;
        pa[6] = (_Float16)a1.z; pa[7] = (_Float16)a1.w;
        float4 b0 = *(const float4*)(gB + k0);
        _Float16* pb = &Bs[brow * LDF + bkg];
        pb[0] = (_Float16)b0.x; pb[1] = (_Float16)b0.y;
        pb[2] = (_Float16)b0.z; pb[3] = (_Float16)b0.w;

        if (kc + 1 < nk) {
            __builtin_prefetch(gA + k0 + 32, 0, 3);
            __builtin_prefetch(gB + k0 + 32, 0, 3);
        }
        __syncthreads();

        FragH fa, fb;
        {
            const int row = mi * 16 + (lane & 15);
            const int kb  = (lane < 16) ? 0 : 16;
            fa.u[0] = *(const uint4*)&As[row * LDF + kb];
            fa.u[1] = *(const uint4*)&As[row * LDF + kb + 8];
        }
        {
            const int row = ni * 16 + (lane & 15);
            const int kb  = (lane < 16) ? 0 : 16;
            fb.u[0] = *(const uint4*)&Bs[row * LDF + kb];
            fb.u[1] = *(const uint4*)&Bs[row * LDF + kb + 8];
        }
        c = __builtin_amdgcn_wmma_f32_16x16x32_f16(false, fa.h, false, fb.h,
                                                   (short)0, c, false, false);
        __syncthreads();
    }

    const int gm_base = m0 + mi * 16 + ((lane < 16) ? 0 : 8);
    const int gn      = n0 + ni * 16 + (lane & 15);
    const float badd  = bias ? bias[gn] : 0.0f;
#pragma unroll
    for (int r = 0; r < 8; ++r) {
        const int gm = gm_base + r;
        size_t orow = (flags & FLAG_REMAP)
                        ? (size_t)((gm & 63) * TDEC + (gm >> 6))
                        : (size_t)gm;
        size_t off = orow * (size_t)N + (size_t)gn;
        float v = c[r] + badd;
        if (flags & FLAG_ACC) v += Y[off];
        Y[off] = v;
    }
}

// GRU gate fusion: h = (1-z)*n + z*h, plus optional f32/f16 snapshots.
__global__ __launch_bounds__(256) void gru_gate(
    const float* __restrict__ gi, const float* __restrict__ gh,
    const float* __restrict__ bias_gi,
    float* __restrict__ h, float* __restrict__ hcopy,
    _Float16* __restrict__ h16, _Float16* __restrict__ hall16)
{
    const int idx = blockIdx.x * blockDim.x + threadIdx.x;
    const int b = idx >> 10;
    const int j = idx & (GRU_H - 1);
    const int base = b * GRU_3H;

    float ir, iz, in_;
    if (bias_gi) {
        ir  = bias_gi[j];
        iz  = bias_gi[GRU_H + j];
        in_ = bias_gi[2 * GRU_H + j];
    } else {
        ir  = gi[base + j];
        iz  = gi[base + GRU_H + j];
        in_ = gi[base + 2 * GRU_H + j];
    }
    const float hr = gh[base + j];
    const float hz = gh[base + GRU_H + j];
    const float hn = gh[base + 2 * GRU_H + j];

    const float r = 1.0f / (1.0f + __expf(-(ir + hr)));
    const float z = 1.0f / (1.0f + __expf(-(iz + hz)));
    const float n = tanhf(in_ + r * hn);
    const float out = (1.0f - z) * n + z * h[idx];
    h[idx] = out;
    if (hcopy)  hcopy[idx]  = out;
    if (h16)    h16[idx]    = (_Float16)out;
    if (hall16) hall16[idx] = (_Float16)out;
}

__global__ __launch_bounds__(256) void gather_words_f32(
    float* __restrict__ words, const float* __restrict__ emb, const int* __restrict__ tgt)
{
    const int idx = blockIdx.x * blockDim.x + threadIdx.x;   // 27*64*512
    const int t = idx >> 15;
    const int b = (idx >> 9) & 63;
    const int j = idx & 511;
    words[idx] = emb[(size_t)tgt[b * 28 + t] * 512 + j];
}

__global__ __launch_bounds__(256) void gather_words_f16(
    _Float16* __restrict__ words, const float* __restrict__ emb, const int* __restrict__ tgt)
{
    const int idx = blockIdx.x * blockDim.x + threadIdx.x;
    const int t = idx >> 15;
    const int b = (idx >> 9) & 63;
    const int j = idx & 511;
    words[idx] = (_Float16)emb[(size_t)tgt[b * 28 + t] * 512 + j];
}

__global__ __launch_bounds__(256) void cvt_f32_to_f16(
    _Float16* __restrict__ dst, const float* __restrict__ src, int n)
{
    const int idx = blockIdx.x * blockDim.x + threadIdx.x;
    if (idx < n) dst[idx] = (_Float16)src[idx];
}

__global__ __launch_bounds__(256) void zero_f32(float* __restrict__ p, int n)
{
    const int idx = blockIdx.x * blockDim.x + threadIdx.x;
    if (idx < n) p[idx] = 0.0f;
}

__global__ __launch_bounds__(256) void zero_f16(_Float16* __restrict__ p, int n)
{
    const int idx = blockIdx.x * blockDim.x + threadIdx.x;
    if (idx < n) p[idx] = (_Float16)0.0f;
}

// In-place row-wise log_softmax over 32000-wide rows.
__global__ __launch_bounds__(256) void logsoftmax_rows(float* __restrict__ out, int ncols)
{
    __shared__ float red[256];
    float* row = out + (size_t)blockIdx.x * (size_t)ncols;
    const int tid = threadIdx.x;

    float m = -INFINITY;
    for (int j = tid; j < ncols; j += 256) m = fmaxf(m, row[j]);
    red[tid] = m; __syncthreads();
    for (int s = 128; s > 0; s >>= 1) {
        if (tid < s) red[tid] = fmaxf(red[tid], red[tid + s]);
        __syncthreads();
    }
    m = red[0]; __syncthreads();

    float sum = 0.0f;
    for (int j = tid; j < ncols; j += 256) sum += __expf(row[j] - m);
    red[tid] = sum; __syncthreads();
    for (int s = 128; s > 0; s >>= 1) {
        if (tid < s) red[tid] += red[tid + s];
        __syncthreads();
    }
    const float shift = m + logf(red[0]);
    for (int j = tid; j < ncols; j += 256) row[j] = row[j] - shift;
}

extern "C" void kernel_launch(void* const* d_in, const int* in_sizes, int n_in,
                              void* d_out, int out_size, void* d_ws, size_t ws_size,
                              hipStream_t stream)
{
    (void)in_sizes; (void)n_in; (void)out_size;

    const float* vid   = (const float*)d_in[0];
    const int*   tgt   = (const int*)  d_in[1];
    const float* emb   = (const float*)d_in[2];
    const float* W_ih1 = (const float*)d_in[3];
    const float* W_hh1 = (const float*)d_in[4];
    const float* b_ih1 = (const float*)d_in[5];
    const float* b_hh1 = (const float*)d_in[6];
    const float* W_ih2 = (const float*)d_in[7];
    const float* W_hh2 = (const float*)d_in[8];
    const float* b_ih2 = (const float*)d_in[9];
    const float* b_hh2 = (const float*)d_in[10];
    const float* W_out = (const float*)d_in[11];
    const float* b_out = (const float*)d_in[12];
    float* out = (float*)d_out;

    const dim3 blk(256);
    const dim3 gRec(GRU_3H / 32, 1);
    const dim3 gGate(64 * GRU_H / 256);

    // f16 fast-path workspace budget
    const size_t NVID = 64 * 40 * 2048, NW1 = 3072 * 2048, NWH1 = 3072 * 1024;
    const size_t NW2 = 3072 * 1536, NWH2 = 3072 * 1024, NWO = (size_t)VOCAB * 1024;
    const size_t NWORDS = TDEC * 64 * 512, NH = 64 * GRU_H, NH2A = TDEC * 64 * GRU_H;
    const size_t F32N = 2 * NH + 2 * 64 * GRU_3H;                     // h1,h2,gi,gh
    const size_t H16N = NVID + NW1 + NWH1 + NW2 + NWH2 + NWO + NWORDS + 2 * NH + NH2A;
    const size_t NEED16 = F32N * 4 + H16N * 2;

    float* ws = (float*)d_ws;
    float* h1 = ws;
    float* h2 = h1 + NH;
    float* gi = h2 + NH;
    float* gh = gi + 64 * GRU_3H;

    if (ws_size >= NEED16) {
        // ---------------- f16 + TDM fast path ----------------
        _Float16* f16b   = (_Float16*)(ws + F32N);
        _Float16* vid16  = f16b;
        _Float16* w1_16  = vid16 + NVID;
        _Float16* wh1_16 = w1_16 + NW1;
        _Float16* w2_16  = wh1_16 + NWH1;
        _Float16* wh2_16 = w2_16 + NW2;
        _Float16* wo_16  = wh2_16 + NWH2;
        _Float16* wrd16  = wo_16 + NWO;
        _Float16* h1_16  = wrd16 + NWORDS;
        _Float16* h2_16  = h1_16 + NH;
        _Float16* h2a16  = h2_16 + NH;

        zero_f32<<<(2 * NH + 255) / 256, blk, 0, stream>>>(h1, 2 * NH);
        zero_f16<<<(2 * NH + 255) / 256, blk, 0, stream>>>(h1_16, 2 * NH);
        cvt_f32_to_f16<<<(NVID + 255) / 256, blk, 0, stream>>>(vid16, vid, NVID);
        cvt_f32_to_f16<<<(NW1 + 255) / 256, blk, 0, stream>>>(w1_16, W_ih1, NW1);
        cvt_f32_to_f16<<<(NWH1 + 255) / 256, blk, 0, stream>>>(wh1_16, W_hh1, NWH1);
        cvt_f32_to_f16<<<(NW2 + 255) / 256, blk, 0, stream>>>(w2_16, W_ih2, NW2);
        cvt_f32_to_f16<<<(NWH2 + 255) / 256, blk, 0, stream>>>(wh2_16, W_hh2, NWH2);
        cvt_f32_to_f16<<<(NWO + 255) / 256, blk, 0, stream>>>(wo_16, W_out, NWO);
        gather_words_f16<<<(NWORDS) / 256, blk, 0, stream>>>(wrd16, emb, tgt);

        for (int t = 0; t < 40; ++t) {
            gemm16<1><<<gRec, blk, 0, stream>>>(vid16 + (size_t)t * 2048, 40 * 2048,
                                                w1_16, 2048, b_ih1, gi, GRU_3H, 2048, 0);
            gemm16<1><<<gRec, blk, 0, stream>>>(h1_16, GRU_H, wh1_16, GRU_H, b_hh1,
                                                gh, GRU_3H, GRU_H, 0);
            gru_gate<<<gGate, blk, 0, stream>>>(gi, gh, nullptr, h1, nullptr, h1_16, nullptr);
            gemm16<1><<<gRec, blk, 0, stream>>>(h1_16, GRU_H, w2_16, 1536, b_ih2,
                                                gi, GRU_3H, GRU_H, 0);
            gemm16<1><<<gRec, blk, 0, stream>>>(h2_16, GRU_H, wh2_16, GRU_H, b_hh2,
                                                gh, GRU_3H, GRU_H, 0);
            gru_gate<<<gGate, blk, 0, stream>>>(gi, gh, nullptr, h2, nullptr, h2_16, nullptr);
        }
        for (int t = 0; t < TDEC; ++t) {
            gemm16<1><<<gRec, blk, 0, stream>>>(h1_16, GRU_H, wh1_16, GRU_H, b_hh1,
                                                gh, GRU_3H, GRU_H, 0);
            gru_gate<<<gGate, blk, 0, stream>>>(nullptr, gh, b_ih1, h1, nullptr, h1_16, nullptr);
            gemm16<1><<<gRec, blk, 0, stream>>>(h1_16, GRU_H, w2_16, 1536, b_ih2,
                                                gi, GRU_3H, GRU_H, 0);
            gemm16<1><<<gRec, blk, 0, stream>>>(wrd16 + (size_t)t * 64 * 512, 512,
                                                w2_16 + 1024, 1536, nullptr,
                                                gi, GRU_3H, 512, FLAG_ACC);
            gemm16<1><<<gRec, blk, 0, stream>>>(h2_16, GRU_H, wh2_16, GRU_H, b_hh2,
                                                gh, GRU_3H, GRU_H, 0);
            gru_gate<<<gGate, blk, 0, stream>>>(gi, gh, nullptr, h2, nullptr, h2_16,
                                                h2a16 + (size_t)t * NH);
        }
        const dim3 gOut(VOCAB / 64, (TDEC * 64) / 64);
        gemm16<2><<<gOut, blk, 0, stream>>>(h2a16, GRU_H, wo_16, GRU_H, b_out,
                                            out, VOCAB, GRU_H, FLAG_REMAP);
    } else {
        // ---------------- f32 fallback path ----------------
        float* words = gh + 64 * GRU_3H;
        float* h2all = words + NWORDS;

        zero_f32<<<(2 * NH + 255) / 256, blk, 0, stream>>>(h1, 2 * NH);
        gather_words_f32<<<(NWORDS) / 256, blk, 0, stream>>>(words, emb, tgt);

        for (int t = 0; t < 40; ++t) {
            gemm_xwT<<<gRec, blk, 0, stream>>>(vid + (size_t)t * 2048, 40 * 2048,
                                               W_ih1, 2048, b_ih1, gi, GRU_3H, 2048, 0);
            gemm_xwT<<<gRec, blk, 0, stream>>>(h1, GRU_H, W_hh1, GRU_H, b_hh1,
                                               gh, GRU_3H, GRU_H, 0);
            gru_gate<<<gGate, blk, 0, stream>>>(gi, gh, nullptr, h1, nullptr, nullptr, nullptr);
            gemm_xwT<<<gRec, blk, 0, stream>>>(h1, GRU_H, W_ih2, 1536, b_ih2,
                                               gi, GRU_3H, GRU_H, 0);
            gemm_xwT<<<gRec, blk, 0, stream>>>(h2, GRU_H, W_hh2, GRU_H, b_hh2,
                                               gh, GRU_3H, GRU_H, 0);
            gru_gate<<<gGate, blk, 0, stream>>>(gi, gh, nullptr, h2, nullptr, nullptr, nullptr);
        }
        for (int t = 0; t < TDEC; ++t) {
            gemm_xwT<<<gRec, blk, 0, stream>>>(h1, GRU_H, W_hh1, GRU_H, b_hh1,
                                               gh, GRU_3H, GRU_H, 0);
            gru_gate<<<gGate, blk, 0, stream>>>(nullptr, gh, b_ih1, h1, nullptr, nullptr, nullptr);
            gemm_xwT<<<gRec, blk, 0, stream>>>(h1, GRU_H, W_ih2, 1536, b_ih2,
                                               gi, GRU_3H, GRU_H, 0);
            gemm_xwT<<<gRec, blk, 0, stream>>>(words + (size_t)t * 64 * 512, 512,
                                               W_ih2 + 1024, 1536, nullptr,
                                               gi, GRU_3H, 512, FLAG_ACC);
            gemm_xwT<<<gRec, blk, 0, stream>>>(h2, GRU_H, W_hh2, GRU_H, b_hh2,
                                               gh, GRU_3H, GRU_H, 0);
            gru_gate<<<gGate, blk, 0, stream>>>(gi, gh, nullptr, h2,
                                                h2all + (size_t)t * NH, nullptr, nullptr);
        }
        const dim3 gOut(VOCAB / 32, (TDEC * 64) / 64);
        gemm_xwT<<<gOut, blk, 0, stream>>>(h2all, GRU_H, W_out, GRU_H, b_out,
                                           out, VOCAB, GRU_H, FLAG_REMAP);
    }
    logsoftmax_rows<<<TDEC * 64, blk, 0, stream>>>(out, VOCAB);
}
